// STLSTMLayer_12154757447799
// MI455X (gfx1250) — matrix operand order repaired and checked by reading, not compiled, not run on hardware
//
#include <hip/hip_runtime.h>
#include <hip/hip_bf16.h>

// ---------------------------------------------------------------------------
// ST-LSTM on gfx1250 (MI455X): bf16 WMMA recurrence, f32 accumulate/state.
// T=64, J=25, B=64, I=256, H=512 (5H=2560; K = I + 2H = 1280)
// ---------------------------------------------------------------------------

typedef unsigned short u16;
typedef __attribute__((ext_vector_type(8)))  float  v8f;
typedef __attribute__((ext_vector_type(16))) __bf16 v16bf;

union FragU { uint4 q[2]; v16bf v; };

static __device__ __forceinline__ u16 f2bf(float f) {
    unsigned int u = __float_as_uint(f);
    unsigned int r = u + 0x7FFFu + ((u >> 16) & 1u);   // round-to-nearest-even
    return (u16)(r >> 16);
}
static __device__ __forceinline__ float sigm(float x) {
    return 1.0f / (1.0f + __expf(-x));
}
static __device__ __forceinline__ float tanh_fast(float x) {
    return fmaf(2.0f, sigm(2.0f * x), -1.0f);
}

// ---------------- prep kernels ----------------

// x (T*J*B*I f32) -> bf16, 4 elements per thread
__global__ __launch_bounds__(256) void k_cvt_x(const float* __restrict__ x,
                                               u16* __restrict__ xbf, int n4) {
    int i4 = (blockIdx.x * 256 + threadIdx.x) * 4;
    if (i4 >= n4) return;
    float4 v = *reinterpret_cast<const float4*>(x + i4);
    ushort4 o;
    o.x = f2bf(v.x); o.y = f2bf(v.y); o.z = f2bf(v.z); o.w = f2bf(v.w);
    *reinterpret_cast<ushort4*>(xbf + i4) = o;
}

// Wt[n][k], n in [0,2560), k in [0,1280): transposed fused [Wx;Whs;Wht] in bf16
__global__ __launch_bounds__(256) void k_build_wt(const float* __restrict__ Wx,
                                                  const float* __restrict__ Whs,
                                                  const float* __restrict__ Wht,
                                                  u16* __restrict__ Wt) {
    int idx = blockIdx.x * 256 + threadIdx.x;
    if (idx >= 2560 * 1280) return;
    int n = idx / 1280;
    int k = idx - n * 1280;
    float v;
    if (k < 256)      v = Wx [(size_t)k         * 2560 + n];
    else if (k < 768) v = Whs[(size_t)(k - 256) * 2560 + n];
    else              v = Wht[(size_t)(k - 768) * 2560 + n];
    Wt[idx] = f2bf(v);
}

// zero h/c state (temporal J*B*H, spatial B*H)
__global__ __launch_bounds__(256) void k_init_state(float* __restrict__ Cs,
                                                    float* __restrict__ Ct,
                                                    u16* __restrict__ Hs,
                                                    u16* __restrict__ HtA) {
    int idx = blockIdx.x * 256 + threadIdx.x;
    if (idx < 25 * 64 * 512) { Ct[idx] = 0.0f; HtA[idx] = 0; }
    if (idx < 64 * 512)      { Cs[idx] = 0.0f; Hs[idx]  = 0; }
}

// ---------------- per-cell GEMM: z = [x(t,j)|h_s|h_t] (64x1280) * Wt^T -> 64x2560
// grid: 80 WGs x 256 thr (8 waves). wave -> one 16x16 output tile.
//   m_tile = wave&3 (4 tiles cover B=64), n_tile = blk*2 + (wave>>2) (160 tiles)
// A staged in LDS in 5 K-chunks of 256 (x | hs.lo | hs.hi | ht.lo | ht.hi).
__global__ __launch_bounds__(256) void k_cell_gemm(const u16* __restrict__ xbf,
                                                   const u16* __restrict__ Wt,
                                                   const u16* __restrict__ Hs,
                                                   const u16* __restrict__ HtA,
                                                   float* __restrict__ zbuf,
                                                   int t, int j) {
    __shared__ u16 sA[64 * 256];            // 32 KB: 64 rows x 256 K-slice (bf16)
    const int tid    = threadIdx.x;
    const int lane   = tid & 31;
    const int wave   = tid >> 5;
    const int m_tile = wave & 3;
    const int n_tile = blockIdx.x * 2 + (wave >> 2);
    const int hiK    = (lane >> 4) & 1;     // lane[4]: selects K half per ISA layout
    const int row_a  = m_tile * 16 + (lane & 15);
    const int col_b  = n_tile * 16 + (lane & 15);

    const u16* xsrc = xbf + (size_t)((t * 25 + j) * 64) * 256;
    const u16* htj  = HtA + (size_t)j * 64 * 512;

    v8f acc = {};

    for (int kc = 0; kc < 5; ++kc) {
        const u16* src;
        int stride;
        if (kc == 0)      { src = xsrc;      stride = 256; }
        else if (kc == 1) { src = Hs;        stride = 512; }
        else if (kc == 2) { src = Hs + 256;  stride = 512; }
        else if (kc == 3) { src = htj;       stride = 512; }
        else              { src = htj + 256; stride = 512; }

        // cooperative 32KB fill: 2048 x 16B chunks over 256 threads
        for (int c = tid; c < 2048; c += 256) {
            int r  = c >> 5;
            int c8 = (c & 31) << 3;
            *reinterpret_cast<uint4*>(&sA[r * 256 + c8]) =
                *reinterpret_cast<const uint4*>(&src[(size_t)r * stride + c8]);
        }
        __syncthreads();

#pragma unroll
        for (int it = 0; it < 8; ++it) {
            FragU a, b;
            // A 16x32 bf16: lane<16 -> K {0..7,16..23}; lane>=16 -> +8
            int ka = it * 32 + hiK * 8;
            a.q[0] = *reinterpret_cast<const uint4*>(&sA[row_a * 256 + ka]);
            a.q[1] = *reinterpret_cast<const uint4*>(&sA[row_a * 256 + ka + 16]);
            // B 32x16 bf16: lane = column, K {0..7,16..23}; lane>=16 -> +8 (same as A)
            size_t kb = (size_t)col_b * 1280 + (size_t)kc * 256 + it * 32 + hiK * 8;
            b.q[0] = *reinterpret_cast<const uint4*>(&Wt[kb]);
            b.q[1] = *reinterpret_cast<const uint4*>(&Wt[kb + 16]);
            acc = __builtin_amdgcn_wmma_f32_16x16x32_bf16(
                false, a.v, false, b.v, (short)0, acc, false, false);
        }
        __syncthreads();
    }

    // C/D layout: VGPR r -> M = r + 8*lane[4], N = lane&15
#pragma unroll
    for (int r = 0; r < 8; ++r) {
        int brow = m_tile * 16 + r + 8 * hiK;
        int col  = n_tile * 16 + (lane & 15);
        zbuf[(size_t)brow * 2560 + col] = acc[r];
    }
}

// ---------------- per-cell gates: c,h update + output ----------------
__global__ __launch_bounds__(256) void k_cell_gates(const float* __restrict__ zbuf,
                                                    const float* __restrict__ bias,
                                                    float* __restrict__ Cs,
                                                    float* __restrict__ Ct,
                                                    u16* __restrict__ Hs,
                                                    u16* __restrict__ HtA,
                                                    float* __restrict__ out,
                                                    int t, int j) {
    int idx = blockIdx.x * 256 + threadIdx.x;      // 0 .. 64*512-1  (b*512+n)
    int n = idx & 511;
    int b = idx >> 9;
    const float* zr = zbuf + (size_t)b * 2560 + n;
    float zi  = zr[0]    + bias[n];
    float zfs = zr[512]  + bias[512  + n];
    float zft = zr[1024] + bias[1024 + n];
    float zg  = zr[1536] + bias[1536 + n];
    float zo  = zr[2048] + bias[2048 + n];

    size_t ji = (size_t)j * (64 * 512) + idx;
    float cs = Cs[idx];
    float ct = Ct[ji];

    float c = sigm(zi) * tanh_fast(zg) + sigm(zfs) * cs + sigm(zft) * ct;
    float h = sigm(zo) * tanh_fast(c);

    Cs[idx] = c;            // running spatial state == deque semantics in (t,j) order
    Ct[ji]  = c;            // temporal state for (t+1, j)
    u16 hb  = f2bf(h);
    Hs[idx] = hb;
    HtA[ji] = hb;
    out[(size_t)(t * 25 + j) * (64 * 512) + idx] = h;
}

// ---------------------------------------------------------------------------

extern "C" void kernel_launch(void* const* d_in, const int* in_sizes, int n_in,
                              void* d_out, int out_size, void* d_ws, size_t ws_size,
                              hipStream_t stream) {
    (void)in_sizes; (void)n_in; (void)out_size; (void)ws_size;
    const float* x   = (const float*)d_in[0];   // (64,25,64,256)
    const float* Wx  = (const float*)d_in[1];   // (256,2560)
    const float* Whs = (const float*)d_in[2];   // (512,2560)
    const float* Wht = (const float*)d_in[3];   // (512,2560)
    const float* bia = (const float*)d_in[4];   // (2560,)
    float* out = (float*)d_out;                 // (64,25,64,512)

    char* ws = (char*)d_ws;
    // workspace layout (all offsets 256B-aligned)
    u16*   xbf  = (u16*)  (ws + 0);             // 52,428,800 B
    u16*   Wt   = (u16*)  (ws + 52428800);      //  6,553,600 B
    float* zbuf = (float*)(ws + 58982400);      //    655,360 B
    u16*   Hs   = (u16*)  (ws + 59637760);      //     65,536 B
    float* Cs   = (float*)(ws + 59703296);      //    131,072 B
    u16*   HtA  = (u16*)  (ws + 59834368);      //  1,638,400 B
    float* Ct   = (float*)(ws + 61472768);      //  3,276,800 B  (end 64,749,568)

    // one-time prep (re-done every call: deterministic)
    k_init_state<<<3200, 256, 0, stream>>>(Cs, Ct, Hs, HtA);
    k_cvt_x<<<25600, 256, 0, stream>>>(x, xbf, 64 * 25 * 64 * 256);
    k_build_wt<<<12800, 256, 0, stream>>>(Wx, Whs, Wht, Wt);

    // serial (t, j) recurrence: 1600 cells, 2 launches each
    for (int t = 0; t < 64; ++t) {
        for (int j = 0; j < 25; ++j) {
            k_cell_gemm<<<80, 256, 0, stream>>>(xbf, Wt, Hs, HtA, zbuf, t, j);
            k_cell_gates<<<128, 256, 0, stream>>>(zbuf, bia, Cs, Ct, Hs, HtA, out, t, j);
        }
    }
}
